// LinAngularAttention_binary_3435973836891
// MI455X (gfx1250) — matrix operand order repaired and checked by reading, not compiled, not run on hardware
//
#include <hip/hip_runtime.h>
#include <stdint.h>

// ---------------------------------------------------------------------------
// LinAngularAttention (binary) for MI455X / gfx1250
//   - all GEMMs on V_WMMA_F32_16X16X32_BF16 (wave32 WMMA)
//   - double-buffered LDS tiles fed by GLOBAL_LOAD_ASYNC_TO_LDS_B128
//     (ASYNCcnt-tracked DMA, no VGPR bounce)
//   - operands stored K-contiguous (B pre-transposed): only 16B-aligned
//     b128 LDS traffic
// ---------------------------------------------------------------------------

typedef unsigned short us;                                    // bf16 storage
typedef __attribute__((ext_vector_type(16))) __bf16 v16bf;    // WMMA A/B frag
typedef __attribute__((ext_vector_type(8)))  float  v8f;      // WMMA C/D frag
typedef int v4i __attribute__((vector_size(16)));             // async-LDS payload

union Frag { v16bf v; uint4 q[2]; };

#if __has_builtin(__builtin_amdgcn_global_load_async_to_lds_b128)
#define USE_ASYNC_LDS 1
#else
#define USE_ASYNC_LDS 0
#endif

__device__ __forceinline__ us f2bf(float f) {
  uint32_t u = __float_as_uint(f);
  uint32_t r = u + 0x7FFFu + ((u >> 16) & 1u);   // round-to-nearest-even
  return (us)(r >> 16);
}

template<int N>
__device__ __forceinline__ void wait_asynccnt() {
#if __has_builtin(__builtin_amdgcn_s_wait_asynccnt)
  __builtin_amdgcn_s_wait_asynccnt(N);
#else
  asm volatile("s_wait_asynccnt %0" :: "i"(N) : "memory");
#endif
}

__device__ __forceinline__ void copy16_to_lds(const us* src, us* dst) {
#if USE_ASYNC_LDS
  // AS1 = flat 64-bit global address; AS3 = low 32 bits of generic LDS addr
  __builtin_amdgcn_global_load_async_to_lds_b128(
      (__attribute__((address_space(1))) v4i*)(unsigned long long)src,
      (__attribute__((address_space(3))) v4i*)(unsigned int)(unsigned long long)dst,
      0, 0);
#else
  *(uint4*)dst = *(const uint4*)src;
#endif
}

// ---------------------------------------------------------------------------
// Tiled WMMA GEMM:  C[M,N] (f32) = A[M,K] (bf16) * Bt[N,K]^T (bf16)
// Bt is B stored transposed (K-contiguous rows). Optional per-column bias.
// Batched via blockIdx.z strides. Double-buffered async LDS pipeline.
// ---------------------------------------------------------------------------
template<int BM, int BN, int BK, int WM, int WN>
__global__ __launch_bounds__(32 * (BM / WM) * (BN / WN))
void gemm_wmma(const us* __restrict__ A, const us* __restrict__ Bt,
               float* __restrict__ C, const float* __restrict__ bias,
               int M, int N, int K,
               long long sA, long long sB, long long sC)
{
  constexpr int LDA = BK + 8;                 // +16B pad keeps 16B alignment
  constexpr int WAVES_N = BN / WN;
  constexpr int NWAVE = (BM / WM) * (BN / WN);
  constexpr int MT = WM / 16, NT = WN / 16;
  constexpr int NTHREADS = 32 * NWAVE;
  constexpr int CH = BK / 8;                  // 16B chunks per tile row
  constexpr int PT = (BM + BN) * CH / NTHREADS; // async ops / thread / tile

  __shared__ us lA[2][BM * LDA];
  __shared__ us lB[2][BN * LDA];

  const long long z = blockIdx.z;
  A  += z * sA;
  Bt += z * sB;
  C  += z * sC;

  const int m0 = blockIdx.y * BM;
  const int n0 = blockIdx.x * BN;
  const int tid  = threadIdx.x;
  const int lane = tid & 31;
  const int wave = tid >> 5;
  const int mW = (wave / WAVES_N) * WM;
  const int nW = (wave % WAVES_N) * WN;

  v8f acc[MT][NT] = {};

  auto stage = [&](int buf, int k0) {
    #pragma unroll
    for (int t = 0; t < PT; ++t) {
      const int i = tid + t * NTHREADS;       // wave-uniform branch: tile
      if (i < BM * CH) {                      // sizes are multiples of 32
        const int r = i / CH, c = (i % CH) * 8;
        copy16_to_lds(&A[(long long)(m0 + r) * K + k0 + c], &lA[buf][r * LDA + c]);
      } else {
        const int j = i - BM * CH;
        const int r = j / CH, c = (j % CH) * 8;
        copy16_to_lds(&Bt[(long long)(n0 + r) * K + k0 + c], &lB[buf][r * LDA + c]);
      }
    }
  };

  const int nk = K / BK;
  stage(0, 0);
  for (int it = 0; it < nk; ++it) {
    const int cur = it & 1;
    const bool more = (it + 1 < nk);
    if (more) stage(cur ^ 1, (it + 1) * BK);  // DMA next tile during compute
#if USE_ASYNC_LDS
    // async loads complete in order: allowing PT outstanding (the tile just
    // issued) guarantees the *current* tile is fully resident in LDS
    if (more) wait_asynccnt<PT>(); else wait_asynccnt<0>();
#endif
    __syncthreads();

    const us* __restrict__ a_s = lA[cur];
    const us* __restrict__ b_s = lB[cur];
    #pragma unroll
    for (int kt = 0; kt < BK; kt += 32) {
      Frag a[MT], b[NT];
      #pragma unroll
      for (int mi = 0; mi < MT; ++mi) {
        // 16-bit A 16x32 layout: lanes 0-15 K=0-7 & 16-23, lanes 16-31 K=8-15 & 24-31
        const us* p = &a_s[(mW + mi * 16 + (lane & 15)) * LDA + kt + ((lane >> 4) * 8)];
        a[mi].q[0] = *(const uint4*)p;
        a[mi].q[1] = *(const uint4*)(p + 16);
      }
      #pragma unroll
      for (int ni = 0; ni < NT; ++ni) {
        // B 32x16 layout: lanes 0-15 K=0-7 & 16-23, lanes 16-31 K=8-15 & 24-31, col = lane&15
        const us* p = &b_s[(nW + ni * 16 + (lane & 15)) * LDA + kt + ((lane >> 4) * 8)];
        b[ni].q[0] = *(const uint4*)p;
        b[ni].q[1] = *(const uint4*)(p + 16);
      }
      #pragma unroll
      for (int mi = 0; mi < MT; ++mi)
        #pragma unroll
        for (int ni = 0; ni < NT; ++ni)
          acc[mi][ni] = __builtin_amdgcn_wmma_f32_16x16x32_bf16(
              false, a[mi].v, false, b[ni].v, (short)0, acc[mi][ni], false, false);
    }
    __syncthreads();
  }

  // ---- C/D layout: VGPR r, lane l -> M = r + 8*(l>>4), Ncol = l&15 ----
  #pragma unroll
  for (int mi = 0; mi < MT; ++mi)
    #pragma unroll
    for (int ni = 0; ni < NT; ++ni) {
      const int ncol = n0 + nW + ni * 16 + (lane & 15);
      const float bv = bias ? bias[ncol] : 0.0f;
      #pragma unroll
      for (int r = 0; r < 8; ++r) {
        const int mrow = m0 + mW + mi * 16 + r + ((lane >> 4) * 8);
        C[(long long)mrow * N + ncol] = acc[mi][ni][r] + bv;
      }
    }
}

// ---------------------------------------------------------------------------
// Elementwise casts
// ---------------------------------------------------------------------------
__global__ void k_cast_bf16(const float* __restrict__ src, us* __restrict__ dst, int n) {
  for (int i = blockIdx.x * blockDim.x + threadIdx.x; i < n; i += gridDim.x * blockDim.x)
    dst[i] = f2bf(src[i]);
}

// src [Kd][Nd] row-major -> dst [Nd][Kd] (bf16), i.e. B^T with K contiguous
__global__ void k_castT_bf16(const float* __restrict__ src, us* __restrict__ dst,
                             int Kd, int Nd) {
  const int total = Kd * Nd;
  for (int i = blockIdx.x * blockDim.x + threadIdx.x; i < total; i += gridDim.x * blockDim.x) {
    int k = i / Nd, n = i % Nd;
    dst[(long long)n * Kd + k] = f2bf(src[i]);
  }
}

__global__ void k_zero4(float* p) { if (threadIdx.x < 4) p[threadIdx.x] = 0.0f; }

// ---------------------------------------------------------------------------
// Per-(b,n,head) L2 norms of q,k + global sums of the normalized values
// ---------------------------------------------------------------------------
__global__ __launch_bounds__(256)
void k_qk_stats(const float* __restrict__ qkv, float* __restrict__ invnq,
                float* __restrict__ invnk, float* __restrict__ gstats)
{
  __shared__ float sred[16];
  const int lane = threadIdx.x & 31;
  const int wave = threadIdx.x >> 5;
  const int gw = blockIdx.x * 8 + wave;
  const int nw = gridDim.x * 8;
  float sq = 0.0f, sk = 0.0f;
  for (int rh = gw; rh < 16384 * 12; rh += nw) {
    const int row = rh / 12, h = rh % 12;
    const float* base = qkv + (long long)row * 2304 + h * 64;
    const float q0 = base[lane],        q1 = base[lane + 32];
    const float k0 = base[768 + lane],  k1 = base[768 + lane + 32];
    float s2q = q0 * q0 + q1 * q1, s2k = k0 * k0 + k1 * k1;
    #pragma unroll
    for (int m = 16; m; m >>= 1) { s2q += __shfl_xor(s2q, m, 32); s2k += __shfl_xor(s2k, m, 32); }
    const float iq = rsqrtf(s2q), ik = rsqrtf(s2k);
    if (lane == 0) { invnq[rh] = iq; invnk[rh] = ik; }
    sq += (q0 + q1) * iq;
    sk += (k0 + k1) * ik;
  }
  #pragma unroll
  for (int m = 16; m; m >>= 1) { sq += __shfl_xor(sq, m, 32); sk += __shfl_xor(sk, m, 32); }
  if (lane == 0) { sred[wave] = sq; sred[8 + wave] = sk; }
  __syncthreads();
  if (threadIdx.x == 0) {
    float aq = 0.0f, ak = 0.0f;
    for (int w = 0; w < 8; ++w) { aq += sred[w]; ak += sred[8 + w]; }
    atomicAdd(&gstats[0], aq);
    atomicAdd(&gstats[1], ak);
  }
}

__global__ void k_finalize_mean(float* g) {
  g[2] = g[0] * (1.0f / 12582912.0f);
  g[3] = g[1] * (1.0f / 12582912.0f);
}

// ---------------------------------------------------------------------------
// Binarize q,k against global means; emit
//   qbin [b,h,n,d] bf16, kT [b,h,d,n] bf16, vT [b,h,e,n] bf16
// ---------------------------------------------------------------------------
__global__ void k_binarize(const float* __restrict__ qkv, const float* __restrict__ invnq,
                           const float* __restrict__ invnk, const float* __restrict__ gstats,
                           us* __restrict__ qbin, us* __restrict__ kT, us* __restrict__ vT)
{
  const float mq = gstats[2], mk = gstats[3];
  const us one = 0x3F80u, zero = 0u;
  for (long long t = blockIdx.x * blockDim.x + threadIdx.x; t < 12582912LL;
       t += (long long)gridDim.x * blockDim.x) {
    const int e = (int)(t & 63);
    const int n = (int)((t >> 6) & 1023);
    const int h = (int)((t >> 16) % 12);
    const int b = (int)(t / 786432);
    const long long row = (long long)b * 1024 + n;
    const int rh = (int)row * 12 + h;
    const float* base = qkv + row * 2304 + h * 64 + e;
    const float qv = base[0]    * invnq[rh];
    const float kv = base[768]  * invnk[rh];
    const float vv = base[1536];
    const long long bh = (long long)b * 12 + h;
    qbin[(bh * 1024 + n) * 64 + e] = (qv - mq > 0.0f) ? one : zero;
    kT  [(bh * 64 + e) * 1024 + n] = (kv - mk > 0.0f) ? one : zero;
    vT  [(bh * 64 + e) * 1024 + n] = f2bf(vv);
  }
}

// attn [z,d,e] f32 -> attnT [z,e,d] bf16 (B^T of q@attn)
__global__ void k_attnT(const float* __restrict__ attn, us* __restrict__ attnT) {
  for (int t = blockIdx.x * blockDim.x + threadIdx.x; t < 192 * 4096;
       t += gridDim.x * blockDim.x) {
    const int z = t >> 12;
    const int e = (t >> 6) & 63;
    const int d = t & 63;
    attnT[t] = f2bf(attn[(z << 12) + (d << 6) + e]);
  }
}

// ---------------------------------------------------------------------------
// Epilogue: out = normalize(0.5*v + (1/pi)*raw) + depthwise_conv9(v)
// ---------------------------------------------------------------------------
__global__ __launch_bounds__(256)
void k_epilogue(const float* __restrict__ raw, const float* __restrict__ qkv,
                const float* __restrict__ wdconv, us* __restrict__ ybf)
{
  const int lane = threadIdx.x & 31;
  const int gw = blockIdx.x * 8 + (threadIdx.x >> 5);
  const int nw = gridDim.x * 8;
  for (int rh = gw; rh < 16384 * 12; rh += nw) {
    const int row = rh / 12, h = rh % 12;      // row = b*1024 + n
    const int b = row >> 10, n = row & 1023;
    const long long bh = (long long)b * 12 + h;
    const float* vbase = qkv + (long long)row * 2304 + 1536 + h * 64;
    const float* rbase = raw + (bh * 1024 + n) * 64;
    float o0 = 0.5f * vbase[lane]      + 0.31830988618379067f * rbase[lane];
    float o1 = 0.5f * vbase[lane + 32] + 0.31830988618379067f * rbase[lane + 32];
    float s2 = o0 * o0 + o1 * o1;
    #pragma unroll
    for (int m = 16; m; m >>= 1) s2 += __shfl_xor(s2, m, 32);
    const float inv = rsqrtf(s2);
    float c0 = 0.0f, c1 = 0.0f;
    #pragma unroll
    for (int j = 0; j < 9; ++j) {
      const int nn = n + j - 4;
      if (nn >= 0 && nn < 1024) {
        const float* vb2 = qkv + ((long long)b * 1024 + nn) * 2304 + 1536 + h * 64;
        const float w = wdconv[h * 9 + j];
        c0 += w * vb2[lane];
        c1 += w * vb2[lane + 32];
      }
    }
    us* yb = ybf + (long long)row * 768 + h * 64;
    yb[lane]      = f2bf(o0 * inv + c0);
    yb[lane + 32] = f2bf(o1 * inv + c1);
  }
}

// ---------------------------------------------------------------------------
extern "C" void kernel_launch(void* const* d_in, const int* in_sizes, int n_in,
                              void* d_out, int out_size, void* d_ws, size_t ws_size,
                              hipStream_t stream)
{
  const float* x      = (const float*)d_in[0];
  // d_in[1]=H, d_in[2]=W (unused by the math)
  const float* w_qkv  = (const float*)d_in[3];
  const float* w_dc   = (const float*)d_in[4];
  const float* w_proj = (const float*)d_in[5];
  const float* b_proj = (const float*)d_in[6];
  float* out = (float*)d_out;
  (void)in_sizes; (void)n_in; (void)out_size; (void)ws_size;

  char* wsp = (char*)d_ws;
  auto alloc = [&](size_t bytes) { char* p = wsp; wsp += (bytes + 255) & ~(size_t)255; return p; };
  us*    x_bf   = (us*)   alloc(12582912ull * 2);   // x bf16
  us*    wqkvT  = (us*)   alloc(1769472ull  * 2);   // w_qkv^T bf16 [2304][768]
  us*    wprojT = (us*)   alloc(589824ull   * 2);   // w_proj^T bf16 [768][768]
  float* qkv    = (float*)alloc(37748736ull * 4);   // [16384][2304] f32
  float* invnq  = (float*)alloc(196608ull   * 4);
  float* invnk  = (float*)alloc(196608ull   * 4);
  float* gstats = (float*)alloc(4 * 4);             // sumq,sumk,meanq,meank
  us*    qbin   = (us*)   alloc(12582912ull * 2);   // [b,h,n,d]
  us*    kT     = (us*)   alloc(12582912ull * 2);   // [b,h,d,n]
  us*    vT     = (us*)   alloc(12582912ull * 2);   // [b,h,e,n]
  float* attn   = (float*)alloc(786432ull   * 4);   // [z,64,64]
  us*    attnT  = (us*)   alloc(786432ull   * 2);   // [z,64,64] transposed bf16
  float* raw    = (float*)alloc(12582912ull * 4);   // q@attn  [b,h,n,e]
  us*    ybf    = (us*)   alloc(12582912ull * 2);   // [B*N,768] bf16

  k_cast_bf16 <<<4096, 256, 0, stream>>>(x, x_bf, 12582912);
  k_castT_bf16<<<2048, 256, 0, stream>>>(w_qkv, wqkvT, 768, 2304);
  k_castT_bf16<<<1024, 256, 0, stream>>>(w_proj, wprojT, 768, 768);
  k_zero4     <<<1,    32,  0, stream>>>(gstats);

  // qkv = x @ w_qkv          (M=16384, N=2304, K=768)  ~58 GFLOP
  gemm_wmma<128, 128, 64, 32, 64><<<dim3(18, 128, 1), 256, 0, stream>>>(
      x_bf, wqkvT, qkv, nullptr, 16384, 2304, 768, 0, 0, 0);

  k_qk_stats     <<<512,  256, 0, stream>>>(qkv, invnq, invnk, gstats);
  k_finalize_mean<<<1,    1,   0, stream>>>(gstats);
  k_binarize     <<<8192, 256, 0, stream>>>(qkv, invnq, invnk, gstats, qbin, kT, vT);

  // attn[z] = kT[z] (64x1024) @ v[z] (1024x64), 192 batches
  gemm_wmma<64, 64, 64, 16, 32><<<dim3(1, 1, 192), 256, 0, stream>>>(
      kT, vT, attn, nullptr, 64, 64, 1024, 65536, 65536, 4096);

  k_attnT<<<1024, 256, 0, stream>>>(attn, attnT);

  // raw[z] = qbin[z] (1024x64) @ attn[z] (64x64), 192 batches
  gemm_wmma<128, 64, 64, 32, 32><<<dim3(1, 8, 192), 256, 0, stream>>>(
      qbin, attnT, raw, nullptr, 1024, 64, 64, 65536, 4096, 65536);

  k_epilogue<<<512, 256, 0, stream>>>(raw, qkv, w_dc, ybf);

  // out = y @ w_proj + b     (M=16384, N=768, K=768)  ~19 GFLOP
  gemm_wmma<128, 128, 64, 32, 64><<<dim3(6, 128, 1), 256, 0, stream>>>(
      ybf, wprojT, out, b_proj, 16384, 768, 768, 0, 0, 0);
}
